// InteractionUnit_2413771620560
// MI455X (gfx1250) — matrix operand order repaired and checked by reading, not compiled, not run on hardware
//
#include <hip/hip_runtime.h>
#include <hip/hip_bf16.h>

typedef __bf16 bf16;
typedef __attribute__((ext_vector_type(16))) __bf16 v16bf;
typedef __attribute__((ext_vector_type(8)))  __bf16 v8bf;
typedef __attribute__((ext_vector_type(8)))  float  v8f;
typedef __attribute__((ext_vector_type(4)))  int    v4i_t;

#define LN_EPS 1e-5f

// ---------------- CDNA5 async global->LDS staging -----------------
#if __has_builtin(__builtin_amdgcn_global_load_async_to_lds_b128)
#define HAVE_ASYNC_LDS 1
typedef __attribute__((address_space(1))) v4i_t av4i_g;
typedef __attribute__((address_space(3))) v4i_t av4i_l;
#endif

__device__ __forceinline__ void copy16_to_lds(const bf16* g, bf16* l) {
#ifdef HAVE_ASYNC_LDS
  __builtin_amdgcn_global_load_async_to_lds_b128(
      (av4i_g*)(bf16*)g, (av4i_l*)l, 0, 0);
#else
  *(uint4*)l = *(const uint4*)g;
#endif
}

__device__ __forceinline__ void wait_async_done() {
#ifdef HAVE_ASYNC_LDS
#if __has_builtin(__builtin_amdgcn_s_wait_asynccnt)
  __builtin_amdgcn_s_wait_asynccnt(0);
#else
  asm volatile("s_wait_asynccnt 0" ::: "memory");
#endif
#endif
}

// ------------------------------------------------------------------
// LayerNorm over last dim R (block = 256 threads, 1 row/block)
// ------------------------------------------------------------------
template<int R, bool OUT_BF16>
__global__ void __launch_bounds__(256)
ln_kernel(const float* __restrict__ in, const float* __restrict__ w,
          const float* __restrict__ b, void* __restrict__ outp) {
  constexpr int T = 256;
  constexpr int PER = R / T;
  const int t = threadIdx.x;
  const size_t row = blockIdx.x;
  const float* xr = in + row * (size_t)R + t * PER;

  float v[PER];
  float s = 0.f, sq = 0.f;
#pragma unroll
  for (int i = 0; i < PER; ++i) {
    v[i] = xr[i];                       // contiguous per thread -> b64/b128
    s  += v[i];
    sq += v[i] * v[i];
  }

  __shared__ float r1[T];
  __shared__ float r2[T];
  r1[t] = s; r2[t] = sq;
  __syncthreads();
#pragma unroll
  for (int off = T / 2; off > 0; off >>= 1) {
    if (t < off) { r1[t] += r1[t + off]; r2[t] += r2[t + off]; }
    __syncthreads();
  }
  const float mu   = r1[0] * (1.0f / R);
  const float var  = r2[0] * (1.0f / R) - mu * mu;
  const float rstd = rsqrtf(var + LN_EPS);

#pragma unroll
  for (int i = 0; i < PER; ++i) {
    const int j = t * PER + i;
    const float y = (v[i] - mu) * rstd * w[j] + b[j];
    if (OUT_BF16) ((bf16*)outp)[row * (size_t)R + j] = (bf16)y;
    else          ((float*)outp)[row * (size_t)R + j] = y;
  }
}

// ------------------------------------------------------------------
// Tiled bf16 transpose: src [R x Cc] -> dst [Cc x R], per batch z
// ------------------------------------------------------------------
__global__ void __launch_bounds__(256)
transpose_kernel(const bf16* __restrict__ src, bf16* __restrict__ dst,
                 int R, int Cc) {
  __shared__ bf16 tile[32][33];
  const int bx = blockIdx.x * 32;   // column base in src
  const int by = blockIdx.y * 32;   // row base in src
  const size_t bs = (size_t)blockIdx.z * (size_t)R * Cc;
  const int tx = threadIdx.x & 31;
  const int ty = threadIdx.x >> 5;  // 0..7
#pragma unroll
  for (int j = 0; j < 32; j += 8)
    tile[ty + j][tx] = src[bs + (size_t)(by + ty + j) * Cc + bx + tx];
  __syncthreads();
#pragma unroll
  for (int j = 0; j < 32; j += 8)
    dst[bs + (size_t)(bx + ty + j) * R + by + tx] = tile[tx][ty + j];
}

// ------------------------------------------------------------------
// bf16 WMMA GEMM, all dims compile-time:
//   C[M x N](f32) = A[M x K] * B[K x N]
// Bsrc row n holds COLUMN n of B (both GEMMs use this after xnT).
// 8 waves as WMW x WNW, wave tile (MT*16) x (NT*16), K-block 64,
// double-buffered LDS staged with async global->LDS loads.
// ------------------------------------------------------------------
#define LP 72   // LDS row pitch in bf16 (144 B = 9*16 B)

template<int M, int N, int K, int LDA, int LDB, int LDC,
         int WMW, int WNW, int MT, int NT>
__global__ void __launch_bounds__(256)
gemm_kernel(const bf16* __restrict__ A, const bf16* __restrict__ Bsrc,
            float* __restrict__ C) {
  static_assert(WMW * WNW == 8, "8 wave32s per workgroup");
  constexpr int TK  = 64;
  constexpr int TMb = WMW * MT * 16;
  constexpr int TNb = WNW * NT * 16;
  constexpr int IA  = TMb * TK / 8 / 256;   // b128 chunks per thread (A)
  constexpr int IB  = TNb * TK / 8 / 256;   // b128 chunks per thread (B)

  __shared__ bf16 As[2][TMb * LP];
  __shared__ bf16 Bs[2][TNb * LP];

  const int t    = threadIdx.x;
  const int lane = t & 31;
  const int wave = t >> 5;
  const int wm   = wave % WMW;
  const int wn   = wave / WMW;
  const int h    = lane >> 4;
  const int l16  = lane & 15;

  const int n0 = blockIdx.x * TNb;
  const int m0 = blockIdx.y * TMb;
  const int bz = blockIdx.z;

  const bf16* Ab = A    + (size_t)bz * (size_t)M * LDA;
  const bf16* Bb = Bsrc + (size_t)bz * (size_t)N * LDB;
  float*      Cb = C    + (size_t)bz * (size_t)M * LDC;

  v8f acc[MT][NT];
  const v8f vzero = {0.f, 0.f, 0.f, 0.f, 0.f, 0.f, 0.f, 0.f};
#pragma unroll
  for (int i = 0; i < MT; ++i)
#pragma unroll
    for (int j = 0; j < NT; ++j) acc[i][j] = vzero;

  // ---- stage one K-block into LDS buffer `buf`
  auto stage = [&](int buf, int kb) {
#pragma unroll
    for (int i = 0; i < IA; ++i) {
      const int idx = t + i * 256;
      const int r  = idx >> 3;           // 8 chunks of 8 bf16 per row
      const int c8 = (idx & 7) << 3;
      copy16_to_lds(Ab + (size_t)(m0 + r) * LDA + kb + c8,
                    &As[buf][r * LP + c8]);
    }
#pragma unroll
    for (int i = 0; i < IB; ++i) {
      const int idx = t + i * 256;
      const int r  = idx >> 3;
      const int c8 = (idx & 7) << 3;
      copy16_to_lds(Bb + (size_t)(n0 + r) * LDB + kb + c8,
                    &Bs[buf][r * LP + c8]);
    }
  };

  stage(0, 0);
  constexpr int NSTEP = K / TK;

  for (int s = 0; s < NSTEP; ++s) {
    const int buf = s & 1;
    wait_async_done();     // current buffer fully landed (our wave's ops)
    __syncthreads();       // ...and everyone else's; prev readers done
    if (s + 1 < NSTEP) stage(buf ^ 1, (s + 1) * TK);   // overlaps compute

#pragma unroll
    for (int ks = 0; ks < 2; ++ks) {
      const int k0 = ks * 32;
      v16bf af[MT];
#pragma unroll
      for (int mt = 0; mt < MT; ++mt) {
        const int r = wm * (MT * 16) + mt * 16 + l16;
        v16bf a;
        ((v8bf*)&a)[0] = *(const v8bf*)&As[buf][r * LP + k0 + h * 8];
        ((v8bf*)&a)[1] = *(const v8bf*)&As[buf][r * LP + k0 + 16 + h * 8];
        af[mt] = a;
      }
#pragma unroll
      for (int nt = 0; nt < NT; ++nt) {
        const int c = wn * (NT * 16) + nt * 16 + l16;
        v16bf bf_;
        ((v8bf*)&bf_)[0] = *(const v8bf*)&Bs[buf][c * LP + k0 + h * 8];
        ((v8bf*)&bf_)[1] = *(const v8bf*)&Bs[buf][c * LP + k0 + 16 + h * 8];
#pragma unroll
        for (int mt = 0; mt < MT; ++mt)
          acc[mt][nt] = __builtin_amdgcn_wmma_f32_16x16x32_bf16(
              false, af[mt], false, bf_, (short)0, acc[mt][nt], false, false);
      }
    }
  }

  // ---- epilogue: M = vgpr + 8*half, N = lane&15; LDC is compile-time so
  // the 8 row stores become immediate-offset global_store_b32.
#pragma unroll
  for (int mt = 0; mt < MT; ++mt) {
#pragma unroll
    for (int nt = 0; nt < NT; ++nt) {
      const int row0 = m0 + wm * (MT * 16) + mt * 16 + h * 8;
      const int col  = n0 + wn * (NT * 16) + nt * 16 + l16;
      float* cp = Cb + (size_t)row0 * LDC + col;
#pragma unroll
      for (int i = 0; i < 8; ++i) cp[i * LDC] = acc[mt][nt][i];
    }
  }
}

// ------------------------------------------------------------------
// Launch: LN -> transpose -> gram GEMM -> LN -> GEMM -> LN
// ------------------------------------------------------------------
extern "C" void kernel_launch(void* const* d_in, const int* in_sizes, int n_in,
                              void* d_out, int out_size, void* d_ws, size_t ws_size,
                              hipStream_t stream) {
  constexpr int B = 4, N = 2048, D = 512;
  const float* x       = (const float*)d_in[0];
  const float* w_token = (const float*)d_in[1];
  const float* b_token = (const float*)d_in[2];
  const float* w_score = (const float*)d_in[3];
  const float* b_score = (const float*)d_in[4];
  float* out = (float*)d_out;

  char* ws = (char*)d_ws;
  const size_t XN_BYTES    = (size_t)B * N * D * sizeof(bf16);   //  8 MB
  const size_t XNT_BYTES   = XN_BYTES;                           //  8 MB
  const size_t SCORE_BYTES = (size_t)B * N * N * sizeof(float);  // 64 MB
  const size_t INTER_BYTES = (size_t)B * N * N * sizeof(bf16);   // 32 MB
  bf16*  xn     = (bf16*)ws;
  bf16*  xnT    = (bf16*)(ws + XN_BYTES);
  float* score  = (float*)(ws + XN_BYTES + XNT_BYTES);
  bf16*  inter  = (bf16*)(ws + XN_BYTES + XNT_BYTES + SCORE_BYTES);
  float* outraw = (float*)(ws + XN_BYTES + XNT_BYTES + SCORE_BYTES + INTER_BYTES);

  // 1) token LayerNorm -> bf16 xn [B][N][D]
  ln_kernel<512, true><<<B * N, 256, 0, stream>>>(x, w_token, b_token, xn);

  // 2) xnT [B][D][N]
  transpose_kernel<<<dim3(D / 32, N / 32, B), 256, 0, stream>>>(xn, xnT, N, D);

  // 3) score = xn @ xn^T  (M=N=2048, K=512); B columns come from xn rows
  gemm_kernel<N, N, D, /*LDA*/D, /*LDB*/D, /*LDC*/N, 2, 4, 4, 2>
      <<<dim3(N / 128, N / 128, B), 256, 0, stream>>>(xn, xn, score);

  // 4) score LayerNorm over N -> bf16 inter
  ln_kernel<2048, true><<<B * N, 256, 0, stream>>>(score, w_score, b_score, inter);

  // 5) out_raw = inter @ xn  (M=2048, N=512, K=2048); B columns = xnT rows
  gemm_kernel<N, D, N, /*LDA*/N, /*LDB*/N, /*LDC*/D, 2, 4, 4, 2>
      <<<dim3(D / 128, N / 128, B), 256, 0, stream>>>(inter, xnT, outraw);

  // 6) final token LayerNorm -> f32 output
  ln_kernel<512, false><<<B * N, 256, 0, stream>>>(outraw, w_token, b_token, out);
}
